// FinePointMatchingOneRef_70695161692712
// MI455X (gfx1250) — matrix operand order repaired and checked by reading, not compiled, not run on hardware
//
#include <hip/hip_runtime.h>
#include <hip/hip_bf16.h>
#include <math.h>

// ---------------------------------------------------------------------------
// Types for CDNA5 WMMA (wave32, v_wmma_f32_16x16x32_bf16)
// ---------------------------------------------------------------------------
typedef __bf16 bf16t;
typedef __attribute__((ext_vector_type(16))) __bf16 v16bf;
typedef __attribute__((ext_vector_type(4)))  __bf16 v4bf;
typedef __attribute__((ext_vector_type(8)))  float  v8f;

// ---------------------------------------------------------------------------
// Model constants
// ---------------------------------------------------------------------------
constexpr int Bn   = 2;
constexpr int Npt  = 2048;   // N1 == N2
constexpr int NTt  = 2049;   // N + bg token
constexpr int NFd  = 512;    // NF
constexpr int NSs  = 513;    // NF + bg token
constexpr int DD   = 256;    // D
constexpr int DHh  = 64;     // D / H
constexpr int Hh   = 4;
constexpr int DFFd = 512;
constexpr int INd  = 512;
constexpr int NBb  = 3;
constexpr long NSQ = (long)NSs * NSs;

// ---------------------------------------------------------------------------
// Generic batched GEMM with bf16 WMMA, fused epilogue
//   C = alpha * op(A) @ op(B) [+ bias[n]] [+ res[m,n]] [relu]
// Batch z: offset = (z/inner)*Outer + (z%inner)*Inner   (for A, B, C, res)
//
// LDS tiles are stored in WMMA *fragment order* so each lane's v16bf is
// 16 contiguous bf16 (two ds_load_b128, no repacking):
//   A 16x32 subtile:  Asf[mSub][lane][e] with lane = (m&15) + 16*hi,
//                     hi = k>>4, e = k & 15
//   B tile (n-major): Bsf[n][k]; lane fragment = Bsf[n][16*hi + e]
// ---------------------------------------------------------------------------
struct GemmParams {
  const float* A; const float* B; float* C;
  const float* bias; const float* res;
  long aO, aI, bO, bI, cO, cI, rO, rI;
  int lda, ldb, ldc, ldr;
  int M, N, K;
  int inner;
  int tA, tB, relu;
  float alpha;
};

__global__ __launch_bounds__(128) void gemm_wmma_k(GemmParams p) {
  __shared__ __align__(16) bf16t Asf[4][32][16];  // 4 KB
  __shared__ __align__(16) bf16t Bsf[64][32];     // 4 KB
  int z  = blockIdx.z;
  int zo = z / p.inner, zi = z % p.inner;
  const float* A  = p.A + (long)zo * p.aO + (long)zi * p.aI;
  const float* Bm = p.B + (long)zo * p.bO + (long)zi * p.bI;
  float*       C  = p.C + (long)zo * p.cO + (long)zi * p.cI;
  const float* R  = p.res ? (p.res + (long)zo * p.rO + (long)zi * p.rI) : nullptr;

  int tileM = blockIdx.y * 64, tileN = blockIdx.x * 64;
  int tid  = threadIdx.x;
  int lane = tid & 31, wave = tid >> 5;
  int wr = (wave >> 1) * 32, wc = (wave & 1) * 32;  // 32x32 per wave
  int mr = lane & 15, hi = lane >> 4;

  v8f acc[2][2] = {};

  for (int k0 = 0; k0 < p.K; k0 += 32) {
    bool fullK = (k0 + 32 <= p.K);
    // ---------------- Stage A tile (64x32) ----------------
    if (!p.tA && ((p.lda & 3) == 0) && fullK && (tileM + 64 <= p.M)) {
      // fast: float4 global loads, packed v4bf LDS stores
#pragma unroll
      for (int q = 0; q < 4; ++q) {
        int qi = tid + q * 128;            // 512 quads
        int m = qi >> 3, kq = (qi & 7) << 2;
        const float4 f4 = *(const float4*)(A + (long)(tileM + m) * p.lda + (k0 + kq));
        int khi = kq >> 4;
        int e0  = kq & 15;
        v4bf pk;
        pk[0] = (bf16t)f4.x; pk[1] = (bf16t)f4.y; pk[2] = (bf16t)f4.z; pk[3] = (bf16t)f4.w;
        *(v4bf*)&Asf[m >> 4][(m & 15) + (khi << 4)][e0] = pk;
      }
      if (k0 + 64 <= p.K)
        __builtin_prefetch(A + (long)(tileM + (tid >> 3)) * p.lda + (k0 + 32 + ((tid & 7) << 2)), 0, 1);
    } else {
      for (int i = tid; i < 64 * 32; i += 128) {
        int m = i >> 5, k = i & 31;
        int gm = tileM + m, gk = k0 + k;
        float v = 0.f;
        if (gm < p.M && gk < p.K)
          v = p.tA ? A[(long)gk * p.lda + gm] : A[(long)gm * p.lda + gk];
        int khi = k >> 4;
        Asf[m >> 4][(m & 15) + (khi << 4)][k & 15] = (bf16t)v;
      }
    }
    // ---------------- Stage B tile (32x64, stored n-major) ----------------
    bool fullB = fullK && (tileN + 64 <= p.N);
    if (fullB && p.tB && ((p.ldb & 3) == 0)) {
      // transB fast: rows of B are K-major -> float4 along k, contiguous LDS
#pragma unroll
      for (int q = 0; q < 4; ++q) {
        int qi = tid + q * 128;
        int n = qi >> 3, kq = (qi & 7) << 2;
        const float4 f4 = *(const float4*)(Bm + (long)(tileN + n) * p.ldb + (k0 + kq));
        v4bf pk;
        pk[0] = (bf16t)f4.x; pk[1] = (bf16t)f4.y; pk[2] = (bf16t)f4.z; pk[3] = (bf16t)f4.w;
        *(v4bf*)&Bsf[n][kq] = pk;
      }
      if (k0 + 64 <= p.K)
        __builtin_prefetch(Bm + (long)(tileN + (tid >> 3)) * p.ldb + (k0 + 32 + ((tid & 7) << 2)), 0, 1);
    } else if (fullB && !p.tB && ((p.ldb & 3) == 0)) {
      // normal fast: float4 along n, 4 scalar LDS stores
#pragma unroll
      for (int q = 0; q < 4; ++q) {
        int qi = tid + q * 128;
        int k = qi >> 4, nq = (qi & 15) << 2;
        const float4 f4 = *(const float4*)(Bm + (long)(k0 + k) * p.ldb + (tileN + nq));
        Bsf[nq + 0][k] = (bf16t)f4.x;
        Bsf[nq + 1][k] = (bf16t)f4.y;
        Bsf[nq + 2][k] = (bf16t)f4.z;
        Bsf[nq + 3][k] = (bf16t)f4.w;
      }
      if (k0 + 64 <= p.K)
        __builtin_prefetch(Bm + (long)(k0 + 32 + (tid >> 4)) * p.ldb + (tileN + ((tid & 15) << 2)), 0, 1);
    } else {
      for (int i = tid; i < 32 * 64; i += 128) {
        int k = i >> 6, n = i & 63;
        int gk = k0 + k, gn = tileN + n;
        float v = 0.f;
        if (gk < p.K && gn < p.N)
          v = p.tB ? Bm[(long)gn * p.ldb + gk] : Bm[(long)gk * p.ldb + gn];
        Bsf[n][k] = (bf16t)v;
      }
    }
    __syncthreads();

    // Fragment loads: contiguous 32B per lane (2 x ds_load_b128 each)
    v16bf fa[2], fb[2];
#pragma unroll
    for (int t = 0; t < 2; ++t) {
      fa[t] = *(const v16bf*)&Asf[(wr >> 4) + t][lane][0];
      fb[t] = *(const v16bf*)&Bsf[wc + t * 16 + mr][hi << 4];
    }
#pragma unroll
    for (int i = 0; i < 2; ++i)
#pragma unroll
      for (int j = 0; j < 2; ++j)
        acc[i][j] = __builtin_amdgcn_wmma_f32_16x16x32_bf16(
            false, fa[i], false, fb[j], (short)0, acc[i][j], false, false);
    __syncthreads();
  }

  // Epilogue: C layout per ISA (m = v + 8*hi, n = lane&15)
#pragma unroll
  for (int i = 0; i < 2; ++i)
#pragma unroll
    for (int j = 0; j < 2; ++j)
#pragma unroll
      for (int v = 0; v < 8; ++v) {
        int gm = tileM + wr + i * 16 + v + 8 * hi;
        int gn = tileN + wc + j * 16 + mr;
        if (gm < p.M && gn < p.N) {
          float val = acc[i][j][v] * p.alpha;
          if (p.bias) val += p.bias[gn];
          if (R)      val += R[(long)gm * p.ldr + gn];
          if (p.relu) val = fmaxf(val, 0.f);
          C[(long)gm * p.ldc + gn] = val;
        }
      }
}

static void run_gemm(hipStream_t st,
                     const float* A, long aO, long aI, int lda, int tA,
                     const float* B, long bO, long bI, int ldb, int tB,
                     float* C, long cO, long cI, int ldc,
                     int M, int N, int K, int batches, int inner, float alpha,
                     const float* bias,
                     const float* res, long rO, long rI, int ldr, int relu) {
  GemmParams p;
  p.A = A; p.B = B; p.C = C; p.bias = bias; p.res = res;
  p.aO = aO; p.aI = aI; p.bO = bO; p.bI = bI; p.cO = cO; p.cI = cI;
  p.rO = rO; p.rI = rI;
  p.lda = lda; p.ldb = ldb; p.ldc = ldc; p.ldr = ldr;
  p.M = M; p.N = N; p.K = K; p.inner = inner;
  p.tA = tA; p.tB = tB; p.relu = relu; p.alpha = alpha;
  dim3 g((N + 63) / 64, (M + 63) / 64, batches);
  gemm_wmma_k<<<g, dim3(128), 0, st>>>(p);
}

// ---------------------------------------------------------------------------
// Ball grouping: first ns indices (ascending) with d2 < r2, pad with first hit
// ---------------------------------------------------------------------------
__global__ void ball_group_k(const float* __restrict__ pts, float* __restrict__ grouped,
                             int ns, float r2) {
  int b = blockIdx.y;
  int i = blockIdx.x * 256 + threadIdx.x;
  if (i >= Npt) return;
  const float* P = pts + (size_t)b * Npt * 3;
  float xi = P[i * 3 + 0], yi = P[i * 3 + 1], zi = P[i * 3 + 2];
  float* G = grouped + ((size_t)b * Npt + i) * ns * 3;
  int cnt = 0; float fx = 0.f, fy = 0.f, fz = 0.f;
  for (int j = 0; j < Npt && cnt < ns; ++j) {
    float dx = P[j * 3 + 0] - xi, dy = P[j * 3 + 1] - yi, dz = P[j * 3 + 2] - zi;
    float d2 = dx * dx + dy * dy + dz * dz;
    if (d2 < r2) {
      if (cnt == 0) { fx = dx; fy = dy; fz = dz; }
      G[cnt * 3 + 0] = dx; G[cnt * 3 + 1] = dy; G[cnt * 3 + 2] = dz;
      ++cnt;
    }
  }
  for (; cnt < ns; ++cnt) { G[cnt * 3 + 0] = fx; G[cnt * 3 + 1] = fy; G[cnt * 3 + 2] = fz; }
}

// ---------------------------------------------------------------------------
// PE shared MLP (3->32->64->128, relu) + max over neighbors. One point/block.
// ---------------------------------------------------------------------------
__global__ __launch_bounds__(128) void pe_mlp_k(
    const float* __restrict__ grouped,
    const float* w0, const float* b0, const float* w1, const float* b1,
    const float* w2, const float* b2,
    float* __restrict__ feats, int ns, int featOff) {
  __shared__ float sw0[96], sb0[32], sw1[2048], sb1[64], sw2[8192], sb2[128];
  __shared__ float hb[64 * 32];
  int tid = threadIdx.x;
  for (int i = tid; i < 96;   i += 128) sw0[i] = w0[i];
  for (int i = tid; i < 32;   i += 128) sb0[i] = b0[i];
  for (int i = tid; i < 2048; i += 128) sw1[i] = w1[i];
  for (int i = tid; i < 64;   i += 128) sb1[i] = b1[i];
  for (int i = tid; i < 8192; i += 128) sw2[i] = w2[i];
  for (int i = tid; i < 128;  i += 128) sb2[i] = b2[i];
  __syncthreads();

  size_t pt = blockIdx.x;
  bool act = tid < ns;
  float h2[64];
  if (act) {
    const float* g = grouped + (pt * ns + tid) * 3;
    float x0 = g[0], x1 = g[1], x2 = g[2];
    float h1[32];
#pragma unroll
    for (int j = 0; j < 32; ++j)
      h1[j] = fmaxf(x0 * sw0[j] + x1 * sw0[32 + j] + x2 * sw0[64 + j] + sb0[j], 0.f);
#pragma unroll
    for (int j = 0; j < 64; ++j) {
      float s = sb1[j];
#pragma unroll
      for (int t2 = 0; t2 < 32; ++t2) s += h1[t2] * sw1[t2 * 64 + j];
      h2[j] = fmaxf(s, 0.f);
    }
  }
  // 4 chunks of 32 output channels; max-reduce over neighbors in LDS
  for (int c = 0; c < 4; ++c) {
    if (act) {
      for (int j = 0; j < 32; ++j) {
        int oc = c * 32 + j;
        float s = sb2[oc];
#pragma unroll
        for (int t2 = 0; t2 < 64; ++t2) s += h2[t2] * sw2[t2 * 128 + oc];
        hb[tid * 32 + j] = fmaxf(s, 0.f);
      }
    }
    __syncthreads();
    if (tid < 32) {
      float m = hb[tid];
      for (int nb = 1; nb < ns; ++nb) m = fmaxf(m, hb[nb * 32 + tid]);
      feats[pt * 256 + featOff + c * 32 + tid] = m;
    }
    __syncthreads();
  }
}

// ---------------------------------------------------------------------------
// Softmax over rows of s (B*H, NSs, NSs) in place, with optional geo bias
// ---------------------------------------------------------------------------
__global__ void softmax_rows_bias(float* __restrict__ s, const float* __restrict__ geo) {
  int row = blockIdx.x, z = blockIdx.y, tid = threadIdx.x;
  int b = z / Hh, h = z % Hh;
  float* S = s + ((size_t)z * NSs + row) * NSs;
  __shared__ float red[256];
  float mx = -1e30f;
  for (int j = tid; j < NSs; j += 256) {
    float v = S[j];
    if (geo && row > 0 && j > 0)
      v += geo[(((size_t)b * NFd + (row - 1)) * NFd + (j - 1)) * Hh + h];
    S[j] = v;
    mx = fmaxf(mx, v);
  }
  red[tid] = mx; __syncthreads();
  for (int st = 128; st > 0; st >>= 1) { if (tid < st) red[tid] = fmaxf(red[tid], red[tid + st]); __syncthreads(); }
  mx = red[0]; __syncthreads();
  float sum = 0.f;
  for (int j = tid; j < NSs; j += 256) { float e = expf(S[j] - mx); S[j] = e; sum += e; }
  red[tid] = sum; __syncthreads();
  for (int st = 128; st > 0; st >>= 1) { if (tid < st) red[tid] += red[tid + st]; __syncthreads(); }
  float inv = 1.f / red[0];
  for (int j = tid; j < NSs; j += 256) S[j] *= inv;
}

// ---------------------------------------------------------------------------
// LayerNorm over D=256 (x already includes residual)
// ---------------------------------------------------------------------------
__global__ void layernorm_k(const float* __restrict__ x, const float* __restrict__ g,
                            const float* __restrict__ b, float* __restrict__ out) {
  int row = blockIdx.x, tid = threadIdx.x;
  __shared__ float red[256];
  size_t base = (size_t)row * 256;
  float v = x[base + tid];
  red[tid] = v; __syncthreads();
  for (int s = 128; s > 0; s >>= 1) { if (tid < s) red[tid] += red[tid + s]; __syncthreads(); }
  float mean = red[0] * (1.f / 256.f);
  __syncthreads();
  float c = v - mean;
  red[tid] = c * c; __syncthreads();
  for (int s = 128; s > 0; s >>= 1) { if (tid < s) red[tid] += red[tid + s]; __syncthreads(); }
  float var = red[0] * (1.f / 256.f);
  out[base + tid] = c * rsqrtf(var + 1e-5f) * g[tid] + b[tid];
}

// ---------------------------------------------------------------------------
// phi(x): relu+1e-6, keep norm, sharpen ^3, renormalize (per 64-dim head)
// ---------------------------------------------------------------------------
__global__ void phi_k(float* __restrict__ x) {
  int row = blockIdx.x, h = blockIdx.y, tid = threadIdx.x;  // 64 threads
  float* p = x + (size_t)row * 256 + h * 64;
  __shared__ float red[64];
  float v = fmaxf(p[tid], 0.f) + 1e-6f;
  red[tid] = v * v; __syncthreads();
  for (int s = 32; s > 0; s >>= 1) { if (tid < s) red[tid] += red[tid + s]; __syncthreads(); }
  float nrm = sqrtf(red[0]); __syncthreads();
  float xp = v * v * v;
  red[tid] = xp * xp; __syncthreads();
  for (int s = 32; s > 0; s >>= 1) { if (tid < s) red[tid] += red[tid + s]; __syncthreads(); }
  float np = sqrtf(red[0]);
  p[tid] = xp / (np + 1e-6f) * nrm;
}

__global__ void ksum_k(const float* __restrict__ kp, float* __restrict__ ksum) {
  int z = blockIdx.x, d = threadIdx.x;  // 64 threads
  int b = z >> 2, h = z & 3;
  float s = 0.f;
  for (int n = 0; n < NSs; ++n) s += kp[((size_t)b * NSs + n) * 256 + h * 64 + d];
  ksum[z * 64 + d] = s;
}

__global__ void zcalc_k(const float* __restrict__ qp, const float* __restrict__ ksum,
                        float* __restrict__ zbf, int total) {
  int i = blockIdx.x * 256 + threadIdx.x;
  if (i >= total) return;
  int h = i & 3, n = (i >> 2) % NTt, b = i / (4 * NTt);
  const float* q = qp + ((size_t)b * NTt + n) * 256 + h * 64;
  const float* ks = ksum + ((size_t)b * 4 + h) * 64;
  float s = 0.f;
  for (int d = 0; d < 64; ++d) s += q[d] * ks[d];
  zbf[i] = 1.f / (s + 1e-6f);
}

__global__ void zscale_k(float* __restrict__ o, const float* __restrict__ zbf, int total) {
  int i = blockIdx.x * 256 + threadIdx.x;
  if (i >= total) return;
  int c = i & 255, row = i >> 8;
  o[i] *= zbf[row * 4 + (c >> 6)];
}

// ---------------------------------------------------------------------------
// Gather sp = [fe[:,0:1], fe[:, 1+idx]]
// ---------------------------------------------------------------------------
__global__ void gather_k(const float* __restrict__ fe, const int* __restrict__ idx,
                         float* __restrict__ sp) {
  int r = blockIdx.x, b = blockIdx.y, c = threadIdx.x;
  int src = (r == 0) ? 0 : 1 + idx[b * NFd + (r - 1)];
  sp[((size_t)b * NSs + r) * 256 + c] = fe[((size_t)b * NTt + src) * 256 + c];
}

__global__ void set_bg_k(float* __restrict__ fe, const float* __restrict__ bg) {
  fe[(size_t)blockIdx.x * NTt * 256 + threadIdx.x] = bg[threadIdx.x];
}

__global__ void set_row0_k(float* __restrict__ fe, const float* __restrict__ c) {
  fe[(size_t)blockIdx.x * NTt * 256 + threadIdx.x] =
      c[(size_t)blockIdx.x * NSs * 256 + threadIdx.x];
}

// ---------------------------------------------------------------------------
// scores, row-normalize, final outputs
// ---------------------------------------------------------------------------
__global__ void score_dot_k(const float* __restrict__ fe, const float* __restrict__ w,
                            const float* __restrict__ sb, float* __restrict__ out, int outOff) {
  int t = blockIdx.x, b = blockIdx.y, tid = threadIdx.x;
  __shared__ float red[256];
  float v = fe[((size_t)b * NTt + t) * 256 + tid] * w[tid];
  red[tid] = v; __syncthreads();
  for (int s = 128; s > 0; s >>= 1) { if (tid < s) red[tid] += red[tid + s]; __syncthreads(); }
  if (tid == 0) out[b * 4098 + outOff + t] = red[0] + sb[2];
}

__global__ void rownorm_k(float* __restrict__ x) {
  int row = blockIdx.x, tid = threadIdx.x;
  __shared__ float red[256];
  size_t base = (size_t)row * 256;
  float v = x[base + tid];
  red[tid] = v * v; __syncthreads();
  for (int s = 128; s > 0; s >>= 1) { if (tid < s) red[tid] += red[tid + s]; __syncthreads(); }
  float n = fmaxf(sqrtf(red[0]), 1e-6f);
  x[base + tid] = v / n;
}

__global__ void score_out_k(const float* __restrict__ scb, float* __restrict__ out) {
  int i = blockIdx.x * 256 + threadIdx.x;
  if (i >= Bn * 4096) return;
  int b = i / 4096, t = i % 4096;
  float s = (t < 2048) ? scb[b * 4098 + 1 + t] : scb[b * 4098 + 2050 + (t - 2048)];
  float sg = 1.f / (1.f + expf(-s));
  out[i] = fminf(fmaxf(sg, 0.f), 1.f);
}

__global__ void sal_row_k(const float* __restrict__ atten, const float* __restrict__ scb,
                          float* __restrict__ sal) {
  int n = blockIdx.x, b = blockIdx.y, tid = threadIdx.x;
  const float* row = atten + ((size_t)b * NTt + 1 + n) * NTt + 1;
  __shared__ float red[256], red2[256];
  float mx = -1e30f;
  for (int m = tid; m < Npt; m += 256) mx = fmaxf(mx, row[m]);
  red[tid] = mx; __syncthreads();
  for (int s = 128; s > 0; s >>= 1) { if (tid < s) red[tid] = fmaxf(red[tid], red[tid + s]); __syncthreads(); }
  mx = red[0]; __syncthreads();
  float sum = 0.f, ws = 0.f;
  for (int m = tid; m < Npt; m += 256) {
    float e = expf(row[m] - mx);
    sum += e; ws += e * scb[b * 4098 + 2050 + m];
  }
  red[tid] = sum; red2[tid] = ws; __syncthreads();
  for (int s = 128; s > 0; s >>= 1) {
    if (tid < s) { red[tid] += red[tid + s]; red2[tid] += red2[tid + s]; }
    __syncthreads();
  }
  if (tid == 0) {
    float v = red2[0] / red[0];
    float sg = 1.f / (1.f + expf(-v));
    sal[(size_t)b * 4096 + n] = fminf(fmaxf(sg, 0.f), 1.f);
  }
}

__global__ void sal_col_k(const float* __restrict__ atten, const float* __restrict__ scb,
                          float* __restrict__ sal) {
  int m = blockIdx.x, b = blockIdx.y, tid = threadIdx.x;
  const float* col = atten + ((size_t)b * NTt + 1) * NTt + 1 + m;  // stride NTt
  __shared__ float red[256], red2[256];
  float mx = -1e30f;
  for (int n = tid; n < Npt; n += 256) mx = fmaxf(mx, col[(size_t)n * NTt]);
  red[tid] = mx; __syncthreads();
  for (int s = 128; s > 0; s >>= 1) { if (tid < s) red[tid] = fmaxf(red[tid], red[tid + s]); __syncthreads(); }
  mx = red[0]; __syncthreads();
  float sum = 0.f, ws = 0.f;
  for (int n = tid; n < Npt; n += 256) {
    float e = expf(col[(size_t)n * NTt] - mx);
    sum += e; ws += e * scb[b * 4098 + 1 + n];
  }
  red[tid] = sum; red2[tid] = ws; __syncthreads();
  for (int s = 128; s > 0; s >>= 1) {
    if (tid < s) { red[tid] += red[tid + s]; red2[tid] += red2[tid + s]; }
    __syncthreads();
  }
  if (tid == 0) {
    float v = red2[0] / red[0];
    float sg = 1.f / (1.f + expf(-v));
    sal[(size_t)b * 4096 + 2048 + m] = fminf(fmaxf(sg, 0.f), 1.f);
  }
}

// ---------------------------------------------------------------------------
// Host orchestration
// ---------------------------------------------------------------------------
extern "C" void kernel_launch(void* const* d_in, const int* in_sizes, int n_in,
                              void* d_out, int out_size, void* d_ws, size_t ws_size,
                              hipStream_t stream) {
  (void)in_sizes; (void)n_in; (void)out_size; (void)ws_size;
  auto FI = [&](int i) { return (const float*)d_in[i]; };
  const float* p1   = FI(0);
  const float* f1in = FI(1);
  const float* geo1 = FI(2);
  const int*   fps1 = (const int*)d_in[3];
  const float* p2   = FI(4);
  const float* f2in = FI(5);
  const float* geo2 = FI(6);
  const int*   fps2 = (const int*)d_in[7];
  // d_in[8] = radius (unused by reference)
  // params: 9..77

  float* out = (float*)d_out;
  float* atten     = out;
  float* score_out = out + (size_t)Bn * NTt * NTt;
  float* sal_out   = score_out + (size_t)Bn * 4096;

  // Workspace carve
  float* ws = (float*)d_ws;
  size_t off = 0;
  auto alloc = [&](size_t n) { float* p = ws + off; off += n; return p; };
  float* f1e = alloc((size_t)Bn * NTt * DD);
  float* f2e = alloc((size_t)Bn * NTt * DD);
  float* sp1 = alloc((size_t)Bn * NSs * DD);
  float* sp2 = alloc((size_t)Bn * NSs * DD);
  float* c1  = alloc((size_t)Bn * NSs * DD);
  float* c2  = alloc((size_t)Bn * NSs * DD);
  float* qp  = alloc((size_t)Bn * NTt * DD);
  float* kbb = alloc((size_t)Bn * NSs * DD);
  float* vbb = alloc((size_t)Bn * NSs * DD);
  float* sbf = alloc((size_t)Bn * Hh * NSQ);
  float* ob  = alloc((size_t)Bn * NTt * DD);
  float* tmp = alloc((size_t)Bn * NTt * DD);
  float* ffh = alloc((size_t)Bn * NTt * DFFd);
  float* grp = alloc((size_t)Bn * Npt * 64 * 3);
  float* pft = alloc((size_t)Bn * NTt * DD);   // PE feats, later fo1
  float* pot = alloc((size_t)Bn * NTt * DD);   // PE proj out, later fo2
  float* ksm = alloc((size_t)Bn * Hh * DHh);
  float* kvm = alloc((size_t)Bn * Hh * DHh * DHh);
  float* zbf = alloc((size_t)Bn * NTt * Hh);
  float* scb = alloc((size_t)Bn * 4098);
  float* fo1 = pft;
  float* fo2 = pot;

  // ---------------- Stage 1: PE + input projection per cloud ---------------
  const float* ptsA[2] = {p1, p2};
  const float* finA[2] = {f1in, f2in};
  float* feA[2] = {f1e, f2e};
  for (int cl = 0; cl < 2; ++cl) {
    const float* pts = ptsA[cl]; const float* fin = finA[cl]; float* fe = feA[cl];
    ball_group_k<<<dim3((Npt + 255) / 256, Bn), 256, 0, stream>>>(pts, grp, 32, 0.09f);
    pe_mlp_k<<<Bn * Npt, 128, 0, stream>>>(grp, FI(18), FI(19), FI(20), FI(21), FI(22), FI(23),
                                           pft, 32, 0);
    ball_group_k<<<dim3((Npt + 255) / 256, Bn), 256, 0, stream>>>(pts, grp, 64, 0.25f);
    pe_mlp_k<<<Bn * Npt, 128, 0, stream>>>(grp, FI(24), FI(25), FI(26), FI(27), FI(28), FI(29),
                                           pft, 64, 128);
    // pe3: feats (B*N,256) @ pe3_w + pe3_b -> pot
    run_gemm(stream, pft, 0, 0, DD, 0, FI(14), 0, 0, DD, 0, pot, 0, 0, DD,
             Bn * Npt, DD, DD, 1, 1, 1.f, FI(15), nullptr, 0, 0, 0, 0);
    // in-proj: fin @ in_w + in_b + pe -> fe rows 1.. (batched over B)
    run_gemm(stream, fin, (long)Npt * INd, 0, INd, 0, FI(9), 0, 0, DD, 0,
             fe + DD, (long)NTt * DD, 0, DD,
             Npt, DD, INd, Bn, 1, 1.f, FI(10), pot, (long)Npt * DD, 0, DD, 0);
    set_bg_k<<<Bn, 256, 0, stream>>>(fe, FI(13));
  }

  // ---------------- Encoder block helpers ----------------------------------
  auto attn = [&](int pb, int blk, float* qin, float* kvin, const float* geo, float* dst) {
    const float* wq = FI(pb + 0) + (size_t)blk * DD * DD;  const float* bq = FI(pb + 1) + blk * DD;
    const float* wk = FI(pb + 2) + (size_t)blk * DD * DD;  const float* bk = FI(pb + 3) + blk * DD;
    const float* wv = FI(pb + 4) + (size_t)blk * DD * DD;  const float* bv = FI(pb + 5) + blk * DD;
    const float* wo = FI(pb + 6) + (size_t)blk * DD * DD;  const float* bo = FI(pb + 7) + blk * DD;
    const float* g1 = FI(pb + 8) + blk * DD;  const float* be1 = FI(pb + 9) + blk * DD;
    const float* g2 = FI(pb + 10) + blk * DD; const float* be2 = FI(pb + 11) + blk * DD;
    const float* fw1 = FI(pb + 12) + (size_t)blk * DD * DFFd; const float* fb1 = FI(pb + 13) + blk * DFFd;
    const float* fw2 = FI(pb + 14) + (size_t)blk * DFFd * DD; const float* fb2 = FI(pb + 15) + blk * DD;
    int Mr = Bn * NSs;
    run_gemm(stream, qin, 0, 0, DD, 0, wq, 0, 0, DD, 0, qp, 0, 0, DD,
             Mr, DD, DD, 1, 1, 1.f, bq, nullptr, 0, 0, 0, 0);
    run_gemm(stream, kvin, 0, 0, DD, 0, wk, 0, 0, DD, 0, kbb, 0, 0, DD,
             Mr, DD, DD, 1, 1, 1.f, bk, nullptr, 0, 0, 0, 0);
    run_gemm(stream, kvin, 0, 0, DD, 0, wv, 0, 0, DD, 0, vbb, 0, 0, DD,
             Mr, DD, DD, 1, 1, 1.f, bv, nullptr, 0, 0, 0, 0);
    // scores per (b,h): Qh @ Kh^T * SCALE
    run_gemm(stream, qp, (long)NSs * DD, DHh, DD, 0, kbb, (long)NSs * DD, DHh, DD, 1,
             sbf, (long)Hh * NSQ, NSQ, NSs,
             NSs, NSs, DHh, Bn * Hh, Hh, 0.125f, nullptr, nullptr, 0, 0, 0, 0);
    softmax_rows_bias<<<dim3(NSs, Bn * Hh), 256, 0, stream>>>(sbf, geo);
    // A @ V
    run_gemm(stream, sbf, (long)Hh * NSQ, NSQ, NSs, 0, vbb, (long)NSs * DD, DHh, DD, 0,
             ob, (long)NSs * DD, DHh, DD,
             NSs, DHh, NSs, Bn * Hh, Hh, 1.f, nullptr, nullptr, 0, 0, 0, 0);
    run_gemm(stream, ob, 0, 0, DD, 0, wo, 0, 0, DD, 0, tmp, 0, 0, DD,
             Mr, DD, DD, 1, 1, 1.f, bo, qin, 0, 0, DD, 0);
    layernorm_k<<<Mr, 256, 0, stream>>>(tmp, g1, be1, dst);
    run_gemm(stream, dst, 0, 0, DD, 0, fw1, 0, 0, DFFd, 0, ffh, 0, 0, DFFd,
             Mr, DFFd, DD, 1, 1, 1.f, fb1, nullptr, 0, 0, 0, 1);
    run_gemm(stream, ffh, 0, 0, DFFd, 0, fw2, 0, 0, DD, 0, tmp, 0, 0, DD,
             Mr, DD, DFFd, 1, 1, 1.f, fb2, dst, 0, 0, DD, 0);
    layernorm_k<<<Mr, 256, 0, stream>>>(tmp, g2, be2, dst);
  };

  auto linattn = [&](int pb, int blk, float* fbig, float* ckv) {
    const float* wq = FI(pb + 0) + (size_t)blk * DD * DD;  const float* bq = FI(pb + 1) + blk * DD;
    const float* wk = FI(pb + 2) + (size_t)blk * DD * DD;  const float* bk = FI(pb + 3) + blk * DD;
    const float* wv = FI(pb + 4) + (size_t)blk * DD * DD;  const float* bv = FI(pb + 5) + blk * DD;
    const float* wo = FI(pb + 6) + (size_t)blk * DD * DD;  const float* bo = FI(pb + 7) + blk * DD;
    const float* g1 = FI(pb + 8) + blk * DD;  const float* be1 = FI(pb + 9) + blk * DD;
    const float* g2 = FI(pb + 10) + blk * DD; const float* be2 = FI(pb + 11) + blk * DD;
    const float* fw1 = FI(pb + 12) + (size_t)blk * DD * DFFd; const float* fb1 = FI(pb + 13) + blk * DFFd;
    const float* fw2 = FI(pb + 14) + (size_t)blk * DFFd * DD; const float* fb2 = FI(pb + 15) + blk * DD;
    int Mq = Bn * NTt, Mk = Bn * NSs;
    run_gemm(stream, fbig, 0, 0, DD, 0, wq, 0, 0, DD, 0, qp, 0, 0, DD,
             Mq, DD, DD, 1, 1, 1.f, bq, nullptr, 0, 0, 0, 0);
    phi_k<<<dim3(Mq, Hh), 64, 0, stream>>>(qp);
    run_gemm(stream, ckv, 0, 0, DD, 0, wk, 0, 0, DD, 0, kbb, 0, 0, DD,
             Mk, DD, DD, 1, 1, 1.f, bk, nullptr, 0, 0, 0, 0);
    phi_k<<<dim3(Mk, Hh), 64, 0, stream>>>(kbb);
    run_gemm(stream, ckv, 0, 0, DD, 0, wv, 0, 0, DD, 0, vbb, 0, 0, DD,
             Mk, DD, DD, 1, 1, 1.f, bv, nullptr, 0, 0, 0, 0);
    ksum_k<<<Bn * Hh, 64, 0, stream>>>(kbb, ksm);
    // kvm = kh^T @ vh  (per b,h): transA
    run_gemm(stream, kbb, (long)NSs * DD, DHh, DD, 1, vbb, (long)NSs * DD, DHh, DD, 0,
             kvm, (long)Hh * DHh * DHh, DHh * DHh, DHh,
             DHh, DHh, NSs, Bn * Hh, Hh, 1.f, nullptr, nullptr, 0, 0, 0, 0);
    int ztot = Bn * NTt * Hh;
    zcalc_k<<<(ztot + 255) / 256, 256, 0, stream>>>(qp, ksm, zbf, ztot);
    // out = qh @ kvm (per b,h)
    run_gemm(stream, qp, (long)NTt * DD, DHh, DD, 0, kvm, (long)Hh * DHh * DHh, DHh * DHh, DHh, 0,
             ob, (long)NTt * DD, DHh, DD,
             NTt, DHh, DHh, Bn * Hh, Hh, 1.f, nullptr, nullptr, 0, 0, 0, 0);
    int stot = Bn * NTt * DD;
    zscale_k<<<(stot + 255) / 256, 256, 0, stream>>>(ob, zbf, stot);
    run_gemm(stream, ob, 0, 0, DD, 0, wo, 0, 0, DD, 0, tmp, 0, 0, DD,
             Mq, DD, DD, 1, 1, 1.f, bo, fbig, 0, 0, DD, 0);
    layernorm_k<<<Mq, 256, 0, stream>>>(tmp, g1, be1, fbig);
    run_gemm(stream, fbig, 0, 0, DD, 0, fw1, 0, 0, DFFd, 0, ffh, 0, 0, DFFd,
             Mq, DFFd, DD, 1, 1, 1.f, fb1, nullptr, 0, 0, 0, 1);
    run_gemm(stream, ffh, 0, 0, DFFd, 0, fw2, 0, 0, DD, 0, tmp, 0, 0, DD,
             Mq, DD, DFFd, 1, 1, 1.f, fb2, fbig, 0, 0, DD, 0);
    layernorm_k<<<Mq, 256, 0, stream>>>(tmp, g2, be2, fbig);
    set_row0_k<<<Bn, 256, 0, stream>>>(fbig, ckv);
  };

  // ---------------- Stage 2: NB encoder iterations --------------------------
  for (int it = 0; it < NBb; ++it) {
    gather_k<<<dim3(NSs, Bn), 256, 0, stream>>>(f1e, fps1, sp1);
    gather_k<<<dim3(NSs, Bn), 256, 0, stream>>>(f2e, fps2, sp2);
    attn(30, it, sp1, sp1, geo1, sp1);     // self-attn cloud 1
    attn(30, it, sp2, sp2, geo2, sp2);     // self-attn cloud 2
    attn(46, it, sp1, sp2, nullptr, c1);   // cross 1<-2
    attn(46, it, sp2, sp1, nullptr, c2);   // cross 2<-1
    linattn(62, it, f1e, c1);              // dense-update cloud 1
    linattn(62, it, f2e, c2);              // dense-update cloud 2
  }

  // ---------------- Stage 3: outputs ----------------------------------------
  score_dot_k<<<dim3(NTt, Bn), 256, 0, stream>>>(f1e, FI(16) + 2 * DD, FI(17), scb, 0);
  score_dot_k<<<dim3(NTt, Bn), 256, 0, stream>>>(f2e, FI(16) + 2 * DD, FI(17), scb, NTt);
  run_gemm(stream, f1e, 0, 0, DD, 0, FI(11), 0, 0, DD, 0, fo1, 0, 0, DD,
           Bn * NTt, DD, DD, 1, 1, 1.f, FI(12), nullptr, 0, 0, 0, 0);
  rownorm_k<<<Bn * NTt, 256, 0, stream>>>(fo1);
  run_gemm(stream, f2e, 0, 0, DD, 0, FI(11), 0, 0, DD, 0, fo2, 0, 0, DD,
           Bn * NTt, DD, DD, 1, 1, 1.f, FI(12), nullptr, 0, 0, 0, 0);
  rownorm_k<<<Bn * NTt, 256, 0, stream>>>(fo2);
  // atten = fo1 @ fo2^T / TEMP
  run_gemm(stream, fo1, (long)NTt * DD, 0, DD, 0, fo2, (long)NTt * DD, 0, DD, 1,
           atten, (long)NTt * NTt, 0, NTt,
           NTt, NTt, DD, Bn, 1, 10.0f, nullptr, nullptr, 0, 0, 0, 0);
  score_out_k<<<(Bn * 4096 + 255) / 256, 256, 0, stream>>>(scb, score_out);
  sal_row_k<<<dim3(Npt, Bn), 256, 0, stream>>>(atten, scb, sal_out);
  sal_col_k<<<dim3(Npt, Bn), 256, 0, stream>>>(atten, scb, sal_out);
}
